// ByteTransformer_88802743812578
// MI455X (gfx1250) — matrix operand order repaired and checked by reading, not compiled, not run on hardware
//
#include <hip/hip_runtime.h>

// ---------------------------------------------------------------------------
// ByteTransformer forward for MI455X (gfx1250, wave32, WMMA bf16).
// Weight-streaming bound (~368MB f32 @ 23.3TB/s ~= 16us floor): every GEMM
// streams f32 weights once, converts to bf16 ONCE (in-register, on the way
// into a double-buffered LDS tile), and feeds v_wmma_f32_16x16x32_bf16 from
// aligned 32B LDS vector loads. Each wave computes 32 output rows so every
// LDS B-fragment feeds two WMMAs; one barrier per K-step, next tile's global
// loads issued before it. Activations stay bf16 between GEMMs; epilogues
// (bias/relu/triple-residual) are fused. Attention is flash-style: one wave
// per (b,h,16 query rows), QK^T and PV via WMMA, online softmax in f32 using
// v_permlane16_b32 16-lane butterflies (DS pipe left free for fragments).
// ---------------------------------------------------------------------------

typedef __bf16 bf16_t;
typedef __attribute__((ext_vector_type(16))) __bf16 v16bf;
typedef __attribute__((ext_vector_type(8)))  __bf16 v8bf;
typedef __attribute__((ext_vector_type(4)))  __bf16 v4bf;
typedef __attribute__((ext_vector_type(8)))  float  v8f;

#define BB   2
#define TT   1024
#define EE   1024
#define MM   4096
#define HH   16
#define CC   64
#define VV   128
#define LL   8
#define BT   2048          // BB*TT
#define F3E  3072          // 3*EE
#define LDB  48            // LDS row stride in bf16 (96B: keeps chunks 16B-aligned)

static __device__ __forceinline__ bf16_t f2bf(float f) { return (bf16_t)f; }

#define SHUF16(lo, hi) \
  __builtin_shufflevector(lo, hi, 0, 1, 2, 3, 4, 5, 6, 7, 8, 9, 10, 11, 12, 13, 14, 15)

// ---- 16-lane butterfly reductions (v_permlane16_b32; both 16-lane groups
// of the wave reduce independently, which matches the WMMA D layout) -------
#if __has_builtin(__builtin_amdgcn_permlane16)
static __device__ __forceinline__ float pl16(float v, unsigned s0, unsigned s1) {
  return __builtin_bit_cast(
      float, __builtin_amdgcn_permlane16(__builtin_bit_cast(unsigned, v),
                                         __builtin_bit_cast(unsigned, v),
                                         s0, s1, false, false));
}
static __device__ __forceinline__ float red16_max(float v) {
  v = fmaxf(v, pl16(v, 0x67452301u, 0xEFCDAB89u));   // xor 1
  v = fmaxf(v, pl16(v, 0x54761032u, 0xDCFE98BAu));   // xor 2
  v = fmaxf(v, pl16(v, 0x32107654u, 0xBA98FEDCu));   // xor 4
  v = fmaxf(v, pl16(v, 0xFEDCBA98u, 0x76543210u));   // xor 8
  return v;
}
static __device__ __forceinline__ float red16_sum(float v) {
  v += pl16(v, 0x67452301u, 0xEFCDAB89u);
  v += pl16(v, 0x54761032u, 0xDCFE98BAu);
  v += pl16(v, 0x32107654u, 0xBA98FEDCu);
  v += pl16(v, 0xFEDCBA98u, 0x76543210u);
  return v;
}
#else
static __device__ __forceinline__ float red16_max(float v) {
#pragma unroll
  for (int m = 8; m >= 1; m >>= 1) v = fmaxf(v, __shfl_xor(v, m, 32));
  return v;
}
static __device__ __forceinline__ float red16_sum(float v) {
#pragma unroll
  for (int m = 8; m >= 1; m >>= 1) v += __shfl_xor(v, m, 32);
  return v;
}
#endif

// ---------------------------------------------------------------------------
// Embedding: x[bt][e] = W_tok[e][byte] + W_pos[e][t]
// ---------------------------------------------------------------------------
__global__ __launch_bounds__(256) void embed_kernel(
    const int* __restrict__ bytes, const float* __restrict__ Wtok,
    const float* __restrict__ Wpos, float* __restrict__ x) {
  const int bt  = blockIdx.x;
  const int t   = bt & (TT - 1);
  const int tok = bytes[bt];
  for (int e = threadIdx.x; e < EE; e += 256)
    x[(size_t)bt * EE + e] = Wtok[(size_t)e * VV + tok] + Wpos[(size_t)e * TT + t];
}

// ---------------------------------------------------------------------------
// LayerNorm over E=1024, one block per row, bf16 output (feeds WMMA A-matrix)
// ---------------------------------------------------------------------------
__global__ __launch_bounds__(256) void ln_kernel(
    const float* __restrict__ x, const float* __restrict__ g,
    const float* __restrict__ bb, bf16_t* __restrict__ h) {
  const int row = blockIdx.x;
  const int tid = threadIdx.x;
  const float* xr = x + (size_t)row * EE;
  float v[4], s = 0.f, q = 0.f;
#pragma unroll
  for (int i = 0; i < 4; ++i) {
    v[i] = xr[tid + i * 256];
    s += v[i];
    q += v[i] * v[i];
  }
#pragma unroll
  for (int m = 16; m >= 1; m >>= 1) {
    s += __shfl_xor(s, m, 32);
    q += __shfl_xor(q, m, 32);
  }
  __shared__ float red[16];
  if ((tid & 31) == 0) { red[tid >> 5] = s; red[8 + (tid >> 5)] = q; }
  __syncthreads();
  float ts = 0.f, tq = 0.f;
#pragma unroll
  for (int w = 0; w < 8; ++w) { ts += red[w]; tq += red[8 + w]; }
  const float mu   = ts * (1.f / (float)EE);
  const float var  = tq * (1.f / (float)EE) - mu * mu;
  const float rstd = rsqrtf(var + 1e-5f);
#pragma unroll
  for (int i = 0; i < 4; ++i) {
    const int e = tid + i * 256;
    h[(size_t)row * EE + e] = f2bf((v[i] - mu) * rstd * g[e] + bb[e]);
  }
}

// ---------------------------------------------------------------------------
// GEMM: out[m][n] = sum_k A_bf16[m][k] * W_f32[n][k]  (+ epilogue)
//   Block: 128 thr = 4 waves; tile BM=128 (32 rows/wave) x BN=128 (8 frags).
//   Double-buffered bf16 weight tile in LDS: next f32 tile is loaded to
//   registers BEFORE the barrier, converted+stored AFTER the WMMA block ->
//   one barrier per K-step and HBM loads hidden under matrix math.
//   Each B-fragment (2x ds_load_b128) feeds two WMMAs (two row-sets).
// EPI: 0 = store f32
//      1 = +bias, relu, store bf16        (MLP up)
//      2 = +bias +res1 +res2, store f32   (MLP down, fused 2x+y+m residual)
//      3 = +bias, store f32               (lm head)
// ---------------------------------------------------------------------------
template <int EPI>
__global__ __launch_bounds__(128) void gemm_bf16_kernel(
    const bf16_t* __restrict__ A, const float* __restrict__ Bw,
    const float* __restrict__ bias, const float* __restrict__ res1,
    const float* __restrict__ res2, void* __restrict__ outp,
    int N, int K) {
  __shared__ bf16_t Bs[2][128 * LDB];          // 2 x 12KB
  const int tid  = threadIdx.x;
  const int lane = tid & 31;
  const int wave = tid >> 5;                   // 0..3
  const int n0 = blockIdx.x * 128;
  const int m0 = blockIdx.y * 128 + wave * 32;

  v8f acc[2][8];
#pragma unroll
  for (int s = 0; s < 2; ++s)
#pragma unroll
    for (int f = 0; f < 8; ++f)
#pragma unroll
      for (int j = 0; j < 8; ++j) acc[s][f][j] = 0.f;

  const int ar0   = m0 + (lane & 15);          // row-set 0
  const int ar1   = ar0 + 16;                  // row-set 1
  const int ahalf = (lane >> 4) << 4;          // A-frag K half: 0 or 16
  const int bklo  = (lane >> 4) << 4;          // B-frag K half: 0 or 16

  // staging: thread tid owns weight row n0+tid (32 f32 per K-step = 128B)
  const float* srow = Bw + (size_t)(n0 + tid) * K;
  float4 w[8];
  {
    const float4* s4 = reinterpret_cast<const float4*>(srow);
#pragma unroll
    for (int i = 0; i < 8; ++i) w[i] = s4[i];
  }

  int buf = 0;
  for (int k0 = 0; k0 < K; k0 += 32) {
    // ---- convert current tile regs -> bf16, store into Bs[buf] ----
    {
      v16bf lo, hi;
#pragma unroll
      for (int i = 0; i < 4; ++i) {
        lo[4 * i + 0] = f2bf(w[i].x);     lo[4 * i + 1] = f2bf(w[i].y);
        lo[4 * i + 2] = f2bf(w[i].z);     lo[4 * i + 3] = f2bf(w[i].w);
        hi[4 * i + 0] = f2bf(w[4 + i].x); hi[4 * i + 1] = f2bf(w[4 + i].y);
        hi[4 * i + 2] = f2bf(w[4 + i].z); hi[4 * i + 3] = f2bf(w[4 + i].w);
      }
      *reinterpret_cast<v16bf*>(&Bs[buf][tid * LDB])      = lo;
      *reinterpret_cast<v16bf*>(&Bs[buf][tid * LDB + 16]) = hi;
    }
    // ---- issue next tile's global loads (hide behind barrier + WMMA) ----
    if (k0 + 32 < K) {
      const float4* s4 = reinterpret_cast<const float4*>(srow + k0 + 32);
#pragma unroll
      for (int i = 0; i < 8; ++i) w[i] = s4[i];
      if (k0 + 64 < K)                         // global_prefetch_b8, tile+2
        __builtin_prefetch(srow + k0 + 64, 0, 1);
    }
    __syncthreads();

    // ---- A fragments (16x32 bf16): chunk [b..b+15], b=0/16 ----
    const bf16_t* ap0 = A + (size_t)ar0 * K + k0 + ahalf;
    const bf16_t* ap1 = A + (size_t)ar1 * K + k0 + ahalf;
    const v16bf af0 = SHUF16(*reinterpret_cast<const v8bf*>(ap0),
                             *reinterpret_cast<const v8bf*>(ap0 + 8));
    const v16bf af1 = SHUF16(*reinterpret_cast<const v8bf*>(ap1),
                             *reinterpret_cast<const v8bf*>(ap1 + 8));

    // ---- 8 N-fragments, each reused for both row-sets ----
#pragma unroll
    for (int f = 0; f < 8; ++f) {
      const v16bf bfrag = *reinterpret_cast<const v16bf*>(
          &Bs[buf][(f * 16 + (lane & 15)) * LDB + bklo]);
      acc[0][f] = __builtin_amdgcn_wmma_f32_16x16x32_bf16(
          false, af0, false, bfrag, (short)0, acc[0][f], false, false);
      acc[1][f] = __builtin_amdgcn_wmma_f32_16x16x32_bf16(
          false, af1, false, bfrag, (short)0, acc[1][f], false, false);
    }
    buf ^= 1;
  }

  // ---- epilogue: D layout row = base + j + 8*(lane>=16), col = lane%16 ----
  const int col = lane & 15;
#pragma unroll
  for (int s = 0; s < 2; ++s) {
    const int rbase = m0 + s * 16 + ((lane >> 4) << 3);
#pragma unroll
    for (int f = 0; f < 8; ++f) {
      const int n = n0 + f * 16 + col;
      const float bv = (EPI != 0) ? bias[n] : 0.f;
#pragma unroll
      for (int j = 0; j < 8; ++j) {
        const size_t off = (size_t)(rbase + j) * N + n;
        float vv = acc[s][f][j] + bv;
        if (EPI == 1) {
          ((bf16_t*)outp)[off] = f2bf(fmaxf(vv, 0.f));
        } else if (EPI == 2) {
          ((float*)outp)[off] = vv + res1[off] + res2[off];
        } else {
          ((float*)outp)[off] = vv;
        }
      }
    }
  }
}

// ---------------------------------------------------------------------------
// Flash attention, one wave per (b, h, 16 query rows). C=64, causal.
//   K tile staged row-major bf16, V tile staged TRANSPOSED bf16 so both
//   B-fragments (QK^T over c; PV over s) are single aligned 32B LDS loads.
//   Online softmax in f32 on the D-register layout, v_permlane16 butterflies.
//   P written to LDS as bf16 for the D->A relayout.
//   out: xb = x + softmax(q k^T / 8) @ v     (first residual fused)
// ---------------------------------------------------------------------------
__global__ __launch_bounds__(32) void attn_kernel(
    const float* __restrict__ qkv, const float* __restrict__ x,
    float* __restrict__ xb) {
  __shared__ bf16_t Ks[32 * LDB];              // [key s][c]        3KB
  __shared__ bf16_t Vt[64 * LDB];              // [c][key s] (T)    6KB
  __shared__ bf16_t Ps[16 * LDB];              // [q row][s]        1.5KB
  const int lane = threadIdx.x;
  const int t0 = blockIdx.x * 16;
  const int b  = blockIdx.y >> 4;
  const int hh = blockIdx.y & 15;
  const size_t hoff = (size_t)hh * (3 * CC);

  const int col    = lane & 15;
  const int rofs   = (lane >> 4) << 3;         // D-layout row offset 0/8
  const int half8  = (lane >> 4) << 3;         // unused; A-frags also use half16 (0/16)
  const int half16 = (lane >> 4) << 4;         // B-frag K base 0/16

  // q fragments (scaled by 1/sqrt(64))
  v16bf qf[2];
  {
    const float* qp = qkv + (size_t)(b * TT + t0 + col) * F3E + hoff;
#pragma unroll
    for (int cc = 0; cc < 2; ++cc) {
      const int cb = cc * 32 + half16;
#pragma unroll
      for (int i = 0; i < 8; ++i) {
        qf[cc][i]     = f2bf(qp[cb + i] * 0.125f);
        qf[cc][i + 8] = f2bf(qp[cb + i + 8] * 0.125f);
      }
    }
  }

  float mrun[8], lrun[8];
  v8f O[4];
#pragma unroll
  for (int j = 0; j < 8; ++j) { mrun[j] = -3.0e38f; lrun[j] = 0.f; }
#pragma unroll
  for (int f = 0; f < 4; ++f)
#pragma unroll
    for (int j = 0; j < 8; ++j) O[f][j] = 0.f;

  for (int s0 = 0; s0 <= t0; s0 += 32) {
    // ---- stage K (row-major) and V (transposed) tiles as bf16 ----
#pragma unroll
    for (int it = 0; it < 16; ++it) {
      const int idx = it * 32 + lane;          // 512 float4-slots
      const int s  = idx >> 4;                 // key row 0..31
      const int c4 = (idx & 15) * 4;           // c chunk 0..60
      const float* rowp = qkv + (size_t)(b * TT + s0 + s) * F3E + hoff;
      const float4 k4 = *reinterpret_cast<const float4*>(rowp + CC + c4);
      const float4 v4 = *reinterpret_cast<const float4*>(rowp + 2 * CC + c4);
      v4bf kb;
      kb[0] = f2bf(k4.x); kb[1] = f2bf(k4.y);
      kb[2] = f2bf(k4.z); kb[3] = f2bf(k4.w);
      *reinterpret_cast<v4bf*>(&Ks[s * LDB + c4]) = kb;
      Vt[(c4 + 0) * LDB + s] = f2bf(v4.x);
      Vt[(c4 + 1) * LDB + s] = f2bf(v4.y);
      Vt[(c4 + 2) * LDB + s] = f2bf(v4.z);
      Vt[(c4 + 3) * LDB + s] = f2bf(v4.w);
    }
    __syncthreads();

    // ---- scores: two 16x16 tiles (keys s0..s0+15, s0+16..s0+31) ----
    v8f S[2];
#pragma unroll
    for (int sh = 0; sh < 2; ++sh)
#pragma unroll
      for (int j = 0; j < 8; ++j) S[sh][j] = 0.f;
#pragma unroll
    for (int sh = 0; sh < 2; ++sh) {
#pragma unroll
      for (int cc = 0; cc < 2; ++cc) {
        const v16bf kf = *reinterpret_cast<const v16bf*>(
            &Ks[(sh * 16 + col) * LDB + cc * 32 + half16]);
        S[sh] = __builtin_amdgcn_wmma_f32_16x16x32_bf16(
            false, qf[cc], false, kf, (short)0, S[sh], false, false);
      }
    }

    // ---- causal mask + online softmax on D layout ----
#pragma unroll
    for (int j = 0; j < 8; ++j) {
      const int t = t0 + rofs + j;
      S[0][j] = (s0 + col > t)      ? -3.0e38f : S[0][j];
      S[1][j] = (s0 + 16 + col > t) ? -3.0e38f : S[1][j];
      const float mx   = red16_max(fmaxf(S[0][j], S[1][j]));
      const float mnew = fmaxf(mrun[j], mx);
      const float p0 = __expf(S[0][j] - mnew);
      const float p1 = __expf(S[1][j] - mnew);
      const float rs = red16_sum(p0 + p1);
      const float scale = __expf(mrun[j] - mnew);
      lrun[j] = lrun[j] * scale + rs;
      mrun[j] = mnew;
      Ps[(rofs + j) * LDB + col]      = f2bf(p0);
      Ps[(rofs + j) * LDB + 16 + col] = f2bf(p1);
#pragma unroll
      for (int f = 0; f < 4; ++f) O[f][j] *= scale;
    }
    __syncthreads();

    // ---- P (A-frag) @ V (B-frag from transposed tile), both 32B loads ----
    const v8bf p0v = *reinterpret_cast<const v8bf*>(&Ps[col * LDB + half16]);
    const v8bf p1v = *reinterpret_cast<const v8bf*>(&Ps[col * LDB + half16 + 8]);
    const v16bf pf = SHUF16(p0v, p1v);
#pragma unroll
    for (int f = 0; f < 4; ++f) {
      const int vc = f * 16 + col;
      const v16bf vf =
          *reinterpret_cast<const v16bf*>(&Vt[vc * LDB + half16]);
      O[f] = __builtin_amdgcn_wmma_f32_16x16x32_bf16(
          false, pf, false, vf, (short)0, O[f], false, false);
    }
    __syncthreads();
  }

  // ---- xb = x + O / l ----
#pragma unroll
  for (int f = 0; f < 4; ++f) {
#pragma unroll
    for (int j = 0; j < 8; ++j) {
      const int t = t0 + rofs + j;
      const size_t off = (size_t)(b * TT + t) * EE + hh * CC + f * 16 + col;
      xb[off] = x[off] + O[f][j] / lrun[j];
    }
  }
}

// ---------------------------------------------------------------------------
// Host-side orchestration (all launches on `stream`; graph-capture safe).
// Workspace layout (60MB total):
//   x 8MB | xb 8MB | h(bf16) 4MB | qkv 24MB | mid(bf16) 16MB
// ---------------------------------------------------------------------------
extern "C" void kernel_launch(void* const* d_in, const int* in_sizes, int n_in,
                              void* d_out, int out_size, void* d_ws,
                              size_t ws_size, hipStream_t stream) {
  (void)in_sizes; (void)n_in; (void)out_size; (void)ws_size;
  const int*   bytes = (const int*)d_in[0];
  const float* W_tok = (const float*)d_in[1];
  const float* W_pos = (const float*)d_in[2];
  const float* Wqkv  = (const float*)d_in[3];
  const float* ln1_g = (const float*)d_in[4];
  const float* ln1_b = (const float*)d_in[5];
  const float* ln2_g = (const float*)d_in[6];
  const float* ln2_b = (const float*)d_in[7];
  const float* W1    = (const float*)d_in[8];
  const float* b1    = (const float*)d_in[9];
  const float* W2    = (const float*)d_in[10];
  const float* b2    = (const float*)d_in[11];
  const float* lnf_g = (const float*)d_in[12];
  const float* lnf_b = (const float*)d_in[13];
  const float* W_out = (const float*)d_in[14];
  const float* b_out = (const float*)d_in[15];

  char* ws = (char*)d_ws;
  float*  x   = (float*)(ws);
  float*  xb  = (float*)(ws + (8ull << 20));
  bf16_t* h   = (bf16_t*)(ws + (16ull << 20));
  float*  qkv = (float*)(ws + (20ull << 20));
  bf16_t* mid = (bf16_t*)(ws + (44ull << 20));

  const dim3 blk256(256), blk128(128), blk32(32);
  const dim3 gQKV(F3E / 128, BT / 128);
  const dim3 gUP (MM  / 128, BT / 128);
  const dim3 gDN (EE  / 128, BT / 128);
  const dim3 gLM (VV  / 128, BT / 128);
  const dim3 gATT(TT / 16, BB * HH);

  embed_kernel<<<BT, blk256, 0, stream>>>(bytes, W_tok, W_pos, x);

  for (int l = 0; l < LL; ++l) {
    const float* Wqkv_l = Wqkv + (size_t)l * F3E * EE;
    const float* W1_l   = W1   + (size_t)l * MM * EE;
    const float* W2_l   = W2   + (size_t)l * EE * MM;

    ln_kernel<<<BT, blk256, 0, stream>>>(x, ln1_g + l * EE, ln1_b + l * EE, h);
    gemm_bf16_kernel<0><<<gQKV, blk128, 0, stream>>>(
        h, Wqkv_l, nullptr, nullptr, nullptr, qkv, F3E, EE);
    attn_kernel<<<gATT, blk32, 0, stream>>>(qkv, x, xb);
    ln_kernel<<<BT, blk256, 0, stream>>>(xb, ln2_g + l * EE, ln2_b + l * EE, h);
    gemm_bf16_kernel<1><<<gUP, blk128, 0, stream>>>(
        h, W1_l, b1 + l * MM, nullptr, nullptr, mid, MM, EE);
    // x_next = x + xb + mlp(xb)  (== 2x + y + m), fused, in-place on x
    gemm_bf16_kernel<2><<<gDN, blk128, 0, stream>>>(
        mid, W2_l, b2 + l * EE, x, xb, x, EE, MM);
  }

  ln_kernel<<<BT, blk256, 0, stream>>>(x, lnf_g, lnf_b, h);
  gemm_bf16_kernel<3><<<gLM, blk128, 0, stream>>>(
      h, W_out, b_out, nullptr, nullptr, d_out, VV, EE);
}
